// Model_78365973283201
// MI455X (gfx1250) — compile-verified, matrix operand order checked by reading, not run on hardware
//
#include <hip/hip_runtime.h>
#include <math.h>

// ---------------- constants ----------------
#define N_SAMP 32768
#define MDIM   128
#define KFB    512
#define BATCH  8
#define NATOMS 16
#define ENVF   128
#define NTF    1024
#define NENV   1024

typedef __attribute__((ext_vector_type(16))) _Float16 v16h;
typedef __attribute__((ext_vector_type(8)))  float    v8f;

// Build a WMMA f16 fragment from LDS: elements 0..7 = p[0..7], 8..15 = p[16..23]
// (wave32 16x16x32 layout: lane half-group hs selects base 8*hs, K-groups 0 and 16)
__device__ __forceinline__ v16h frag16(const _Float16* __restrict__ p) {
  v16h f;
#pragma unroll
  for (int i = 0; i < 8; ++i) { f[i] = p[i]; f[i + 8] = p[i + 16]; }
  return f;
}

// clamp + zero-mask (branchless; load address always valid)
__device__ __forceinline__ _Float16 ld_masked(const float* __restrict__ s, int t) {
  int tc = t < 0 ? 0 : (t >= N_SAMP ? N_SAMP - 1 : t);
  float v = s[tc];
  return ((unsigned)t < (unsigned)N_SAMP) ? (_Float16)v : (_Float16)0.f;
}

// ============================================================
// 1) Filterbank conv as GEMM: h[b,o,t] = sum_k fb[o,k] * x[b, t+k-256]
//    LDS-staged: A chunk 128x32, B chunk 32x32 (f16)
// ============================================================
__global__ __launch_bounds__(256) void k_fbconv(const float* __restrict__ x,
                                                const float* __restrict__ fb,
                                                float* __restrict__ h) {
  __shared__ _Float16 sA[MDIM][32];
  __shared__ _Float16 sB[32][32];
  const int tid = threadIdx.x;
  const int lane = tid & 31, wave = tid >> 5;
  const int l16 = lane & 15, hs = lane >> 4;
  const int colBase = blockIdx.x * 32;          // over B*T
  const int b = colBase >> 15;
  const int tBase = colBase & (N_SAMP - 1);
  const int m0 = wave * 16;
  const float* xb = x + (size_t)b * N_SAMP;
  // staging coords
  const int am = tid >> 1, ak0 = (tid & 1) * 16;          // A: 16 floats / thread
  const int bk = tid >> 3, bn0 = (tid & 7) * 4;           // B: 4 elems / thread
  v8f acc0 = {}, acc1 = {};
  for (int kc = 0; kc < KFB; kc += 32) {
    // stage A (coalesced float4 x4)
    {
      const float4* s4 = (const float4*)(fb + (size_t)am * KFB + kc + ak0);
      float4 v0 = s4[0], v1 = s4[1], v2 = s4[2], v3 = s4[3];
      _Float16* d = &sA[am][ak0];
      d[0]=(_Float16)v0.x; d[1]=(_Float16)v0.y; d[2]=(_Float16)v0.z; d[3]=(_Float16)v0.w;
      d[4]=(_Float16)v1.x; d[5]=(_Float16)v1.y; d[6]=(_Float16)v1.z; d[7]=(_Float16)v1.w;
      d[8]=(_Float16)v2.x; d[9]=(_Float16)v2.y; d[10]=(_Float16)v2.z; d[11]=(_Float16)v2.w;
      d[12]=(_Float16)v3.x; d[13]=(_Float16)v3.y; d[14]=(_Float16)v3.z; d[15]=(_Float16)v3.w;
    }
    // stage B (branchless clamp+mask)
    {
      const int kg = kc + bk;
      const int tg = tBase + bn0 + kg - 256;
#pragma unroll
      for (int q = 0; q < 4; ++q) sB[bn0 + q][bk] = ld_masked(xb, tg + q);
      if (kc + 32 < KFB) __builtin_prefetch(&xb[(tg + 32) < 0 ? 0 : (tg + 32)]);
    }
    __syncthreads();
    v16h a  = frag16(&sA[m0 + l16][8 * hs]);
    v16h b0 = frag16(&sB[l16][8 * hs]);
    v16h b1 = frag16(&sB[16 + l16][8 * hs]);
    acc0 = __builtin_amdgcn_wmma_f32_16x16x32_f16(false, a, false, b0, (short)0, acc0, false, false);
    acc1 = __builtin_amdgcn_wmma_f32_16x16x32_f16(false, a, false, b1, (short)0, acc1, false, false);
    __syncthreads();
  }
#pragma unroll
  for (int r = 0; r < 8; ++r) {
    const int m = m0 + r + 8 * hs;
    const size_t base = ((size_t)b * MDIM + m) * N_SAMP + tBase;
    h[base + l16]      = acc0[r];
    h[base + 16 + l16] = acc1[r];
  }
}

// ============================================================
// 2) Dilated conv (K=3, dilation d) as GEMM over K=384 im2col
// ============================================================
__global__ __launch_bounds__(256) void k_dil(const float* __restrict__ hin,
                                             const float* __restrict__ Wd,
                                             const float* __restrict__ bias,
                                             float* __restrict__ y, int d) {
  __shared__ _Float16 sA[MDIM][32];
  __shared__ _Float16 sB[32][32];
  const int tid = threadIdx.x;
  const int lane = tid & 31, wave = tid >> 5;
  const int l16 = lane & 15, hs = lane >> 4;
  const int colBase = blockIdx.x * 32;
  const int b = colBase >> 15;
  const int tBase = colBase & (N_SAMP - 1);
  const int m0 = wave * 16;
  const float* hb = hin + (size_t)b * MDIM * N_SAMP;
  const int am = tid >> 1, ak0 = (tid & 1) * 16;
  const int bk = tid >> 3, bn0 = (tid & 7) * 4;
  v8f acc0 = {}, acc1 = {};
  for (int kc = 0; kc < 384; kc += 32) {
    {
      const float4* s4 = (const float4*)(Wd + (size_t)am * 384 + kc + ak0);
      float4 v0 = s4[0], v1 = s4[1], v2 = s4[2], v3 = s4[3];
      _Float16* dd = &sA[am][ak0];
      dd[0]=(_Float16)v0.x; dd[1]=(_Float16)v0.y; dd[2]=(_Float16)v0.z; dd[3]=(_Float16)v0.w;
      dd[4]=(_Float16)v1.x; dd[5]=(_Float16)v1.y; dd[6]=(_Float16)v1.z; dd[7]=(_Float16)v1.w;
      dd[8]=(_Float16)v2.x; dd[9]=(_Float16)v2.y; dd[10]=(_Float16)v2.z; dd[11]=(_Float16)v2.w;
      dd[12]=(_Float16)v3.x; dd[13]=(_Float16)v3.y; dd[14]=(_Float16)v3.z; dd[15]=(_Float16)v3.w;
    }
    {
      const int kg = kc + bk;         // k = c*3 + j
      const int c  = kg / 3;
      const int j  = kg - 3 * c;
      const int tg = tBase + bn0 + (j - 1) * d;
      const float* src = hb + (size_t)c * N_SAMP;
#pragma unroll
      for (int q = 0; q < 4; ++q) sB[bn0 + q][bk] = ld_masked(src, tg + q);
      if (kc + 32 < 384) __builtin_prefetch(&src[tg < 0 ? 0 : tg]);
    }
    __syncthreads();
    v16h a  = frag16(&sA[m0 + l16][8 * hs]);
    v16h b0 = frag16(&sB[l16][8 * hs]);
    v16h b1 = frag16(&sB[16 + l16][8 * hs]);
    acc0 = __builtin_amdgcn_wmma_f32_16x16x32_f16(false, a, false, b0, (short)0, acc0, false, false);
    acc1 = __builtin_amdgcn_wmma_f32_16x16x32_f16(false, a, false, b1, (short)0, acc1, false, false);
    __syncthreads();
  }
#pragma unroll
  for (int r = 0; r < 8; ++r) {
    const int m = m0 + r + 8 * hs;
    const size_t base = ((size_t)b * MDIM + m) * N_SAMP + tBase;
    y[base + l16]      = acc0[r] + bias[m];
    y[base + 16 + l16] = acc1[r] + bias[m];
  }
}

// ============================================================
// 3) Pointwise GEMM + bias + residual
// ============================================================
__global__ __launch_bounds__(256) void k_pw(const float* __restrict__ yin,
                                            const float* __restrict__ hres,
                                            const float* __restrict__ Wp,
                                            const float* __restrict__ bias,
                                            float* __restrict__ hout) {
  __shared__ _Float16 sA[MDIM][32];
  __shared__ _Float16 sB[32][32];
  const int tid = threadIdx.x;
  const int lane = tid & 31, wave = tid >> 5;
  const int l16 = lane & 15, hs = lane >> 4;
  const int colBase = blockIdx.x * 32;
  const int b = colBase >> 15;
  const int tBase = colBase & (N_SAMP - 1);
  const int m0 = wave * 16;
  const float* yb = yin + (size_t)b * MDIM * N_SAMP;
  const int am = tid >> 1, ak0 = (tid & 1) * 16;
  const int bk = tid >> 3, bn0 = (tid & 7) * 4;
  v8f acc0 = {}, acc1 = {};
#pragma unroll
  for (int kc = 0; kc < MDIM; kc += 32) {
    {
      const float4* s4 = (const float4*)(Wp + (size_t)am * MDIM + kc + ak0);
      float4 v0 = s4[0], v1 = s4[1], v2 = s4[2], v3 = s4[3];
      _Float16* dd = &sA[am][ak0];
      dd[0]=(_Float16)v0.x; dd[1]=(_Float16)v0.y; dd[2]=(_Float16)v0.z; dd[3]=(_Float16)v0.w;
      dd[4]=(_Float16)v1.x; dd[5]=(_Float16)v1.y; dd[6]=(_Float16)v1.z; dd[7]=(_Float16)v1.w;
      dd[8]=(_Float16)v2.x; dd[9]=(_Float16)v2.y; dd[10]=(_Float16)v2.z; dd[11]=(_Float16)v2.w;
      dd[12]=(_Float16)v3.x; dd[13]=(_Float16)v3.y; dd[14]=(_Float16)v3.z; dd[15]=(_Float16)v3.w;
    }
    {
      const float4 v = *(const float4*)(yb + (size_t)(kc + bk) * N_SAMP + tBase + bn0);
      sB[bn0 + 0][bk] = (_Float16)v.x;
      sB[bn0 + 1][bk] = (_Float16)v.y;
      sB[bn0 + 2][bk] = (_Float16)v.z;
      sB[bn0 + 3][bk] = (_Float16)v.w;
      if (kc + 32 < MDIM) __builtin_prefetch(yb + (size_t)(kc + 32 + bk) * N_SAMP + tBase + bn0);
    }
    __syncthreads();
    v16h a  = frag16(&sA[m0 + l16][8 * hs]);
    v16h b0 = frag16(&sB[l16][8 * hs]);
    v16h b1 = frag16(&sB[16 + l16][8 * hs]);
    acc0 = __builtin_amdgcn_wmma_f32_16x16x32_f16(false, a, false, b0, (short)0, acc0, false, false);
    acc1 = __builtin_amdgcn_wmma_f32_16x16x32_f16(false, a, false, b1, (short)0, acc1, false, false);
    __syncthreads();
  }
#pragma unroll
  for (int r = 0; r < 8; ++r) {
    const int m = m0 + r + 8 * hs;
    const size_t base = ((size_t)b * MDIM + m) * N_SAMP + tBase;
    hout[base + l16]      = acc0[r] + bias[m] + hres[base + l16];
    hout[base + 16 + l16] = acc1[r] + bias[m] + hres[base + 16 + l16];
  }
}

// ============================================================
// 4) Generic linear (128 rows x 128 in) -> Nout, optional leaky
// ============================================================
__global__ __launch_bounds__(256) void k_linear(const float* __restrict__ V,
                                                const float* __restrict__ Wt,
                                                const float* __restrict__ bias,
                                                float* __restrict__ out,
                                                int Nout, int leaky) {
  __shared__ _Float16 sA[MDIM][32];
  __shared__ _Float16 sB[32][32];
  const int tid = threadIdx.x;
  const int lane = tid & 31, wave = tid >> 5;
  const int l16 = lane & 15, hs = lane >> 4;
  const int nBase = blockIdx.x * 32;
  const int m0 = wave * 16;
  const int am = tid >> 1, ak0 = (tid & 1) * 16;
  const int bn = tid >> 3, bk0 = (tid & 7) * 4;   // B: n fixed, 4 contiguous k
  v8f acc0 = {}, acc1 = {};
#pragma unroll
  for (int kc = 0; kc < MDIM; kc += 32) {
    {
      const float4* s4 = (const float4*)(V + (size_t)am * MDIM + kc + ak0);
      float4 v0 = s4[0], v1 = s4[1], v2 = s4[2], v3 = s4[3];
      _Float16* dd = &sA[am][ak0];
      dd[0]=(_Float16)v0.x; dd[1]=(_Float16)v0.y; dd[2]=(_Float16)v0.z; dd[3]=(_Float16)v0.w;
      dd[4]=(_Float16)v1.x; dd[5]=(_Float16)v1.y; dd[6]=(_Float16)v1.z; dd[7]=(_Float16)v1.w;
      dd[8]=(_Float16)v2.x; dd[9]=(_Float16)v2.y; dd[10]=(_Float16)v2.z; dd[11]=(_Float16)v2.w;
      dd[12]=(_Float16)v3.x; dd[13]=(_Float16)v3.y; dd[14]=(_Float16)v3.z; dd[15]=(_Float16)v3.w;
    }
    {
      // B[k][n] = Wt[nBase+n][k]; load 4 contiguous k per thread
      const float4 v = *(const float4*)(Wt + (size_t)(nBase + bn) * MDIM + kc + bk0);
      _Float16* dd = &sB[bn][bk0];
      dd[0] = (_Float16)v.x; dd[1] = (_Float16)v.y; dd[2] = (_Float16)v.z; dd[3] = (_Float16)v.w;
    }
    __syncthreads();
    v16h a  = frag16(&sA[m0 + l16][8 * hs]);
    v16h b0 = frag16(&sB[l16][8 * hs]);
    v16h b1 = frag16(&sB[16 + l16][8 * hs]);
    acc0 = __builtin_amdgcn_wmma_f32_16x16x32_f16(false, a, false, b0, (short)0, acc0, false, false);
    acc1 = __builtin_amdgcn_wmma_f32_16x16x32_f16(false, a, false, b1, (short)0, acc1, false, false);
    __syncthreads();
  }
#pragma unroll
  for (int r = 0; r < 8; ++r) {
    const int m = m0 + r + 8 * hs;
    const int o0 = nBase + l16, o1 = nBase + 16 + l16;
    float v0 = acc0[r] + bias[o0];
    float v1 = acc1[r] + bias[o1];
    if (leaky) { if (v0 < 0.f) v0 *= 0.2f; if (v1 < 0.f) v1 *= 0.2f; }
    out[(size_t)m * Nout + o0] = v0;
    out[(size_t)m * Nout + o1] = v1;
  }
}

// ============================================================
// 5) attention logits: attn[b,t] = sum_c aw[c]*h[b,c,t] + ab
// ============================================================
__global__ __launch_bounds__(256) void k_attn(const float* __restrict__ h,
                                              const float* __restrict__ aw,
                                              const float* __restrict__ ab,
                                              float* __restrict__ logits) {
  const size_t i = (size_t)blockIdx.x * 256 + threadIdx.x;  // over B*T
  const int b = (int)(i >> 15);
  const int t = (int)(i & (N_SAMP - 1));
  float s = ab[0];
  const float* hb = h + (size_t)b * MDIM * N_SAMP + t;
#pragma unroll 4
  for (int c = 0; c < MDIM; ++c) s += aw[c] * hb[(size_t)c * N_SAMP];
  logits[i] = s;
}

// ============================================================
// 6) softmax + top-16 (descending, tie -> lowest index)
// ============================================================
__global__ __launch_bounds__(256) void k_topk(const float* __restrict__ logits,
                                              int* __restrict__ oidx,
                                              float* __restrict__ oval) {
  const int b = blockIdx.x, tid = threadIdx.x;
  const float* L = logits + (size_t)b * N_SAMP;
  __shared__ float sv[256];
  __shared__ int   si[256];
  __shared__ int   sel[NATOMS];
  __shared__ float sh_mx, sh_sum;
  float mx = -3.4e38f;
  for (int i = tid; i < N_SAMP; i += 256) mx = fmaxf(mx, L[i]);
  sv[tid] = mx; __syncthreads();
  for (int st = 128; st > 0; st >>= 1) { if (tid < st) sv[tid] = fmaxf(sv[tid], sv[tid + st]); __syncthreads(); }
  if (tid == 0) sh_mx = sv[0];
  __syncthreads();
  const float M = sh_mx;
  float sm = 0.f;
  for (int i = tid; i < N_SAMP; i += 256) sm += expf(L[i] - M);
  sv[tid] = sm; __syncthreads();
  for (int st = 128; st > 0; st >>= 1) { if (tid < st) sv[tid] += sv[tid + st]; __syncthreads(); }
  if (tid == 0) sh_sum = sv[0];
  __syncthreads();
  const float S = sh_sum;
  for (int a = 0; a < NATOMS; ++a) {
    float best = -3.4e38f; int bi = 0x7fffffff;
    for (int i = tid; i < N_SAMP; i += 256) {
      bool skip = false;
      for (int j = 0; j < a; ++j) if (sel[j] == i) skip = true;
      if (skip) continue;
      const float v = L[i];
      if (v > best || (v == best && i < bi)) { best = v; bi = i; }
    }
    sv[tid] = best; si[tid] = bi; __syncthreads();
    for (int st = 128; st > 0; st >>= 1) {
      if (tid < st) {
        if (sv[tid + st] > sv[tid] || (sv[tid + st] == sv[tid] && si[tid + st] < si[tid])) {
          sv[tid] = sv[tid + st]; si[tid] = si[tid + st];
        }
      }
      __syncthreads();
    }
    if (tid == 0) {
      sel[a] = si[0];
      oidx[b * NATOMS + a] = si[0];
      oval[b * NATOMS + a] = expf(sv[0] - M) / S;
    }
    __syncthreads();
  }
}

// 7) gather latents: z[r][c] = h[b,c,idx[r]] * val[r]
__global__ void k_gatherz(const float* __restrict__ h, const int* __restrict__ idx,
                          const float* __restrict__ val, float* __restrict__ z) {
  const int r = blockIdx.x, c = threadIdx.x;
  const int b = r >> 4;
  const int t = idx[r];
  z[(size_t)r * MDIM + c] = h[((size_t)b * MDIM + c) * N_SAMP + t] * val[r];
}

// 8) row argmax over n (tie -> lowest index)
__global__ __launch_bounds__(256) void k_argmax(const float* __restrict__ L, int n,
                                                int* __restrict__ outIdx) {
  const int r = blockIdx.x, tid = threadIdx.x;
  const float* row = L + (size_t)r * n;
  __shared__ float sv[256]; __shared__ int si[256];
  float best = -3.4e38f; int bi = 0x7fffffff;
  for (int i = tid; i < n; i += 256) {
    const float v = row[i];
    if (v > best || (v == best && i < bi)) { best = v; bi = i; }
  }
  sv[tid] = best; si[tid] = bi; __syncthreads();
  for (int st = 128; st > 0; st >>= 1) {
    if (tid < st) {
      if (sv[tid + st] > sv[tid] || (sv[tid + st] == sv[tid] && si[tid + st] < si[tid])) {
        sv[tid] = sv[tid + st]; si[tid] = si[tid + st];
      }
    }
    __syncthreads();
  }
  if (tid == 0) outIdx[r] = si[0];
}

// 9) amp head: amp[r] = |sum_c V[r][c]*ow[c] + ob|
__global__ void k_amp(const float* __restrict__ V, const float* __restrict__ ow,
                      const float* __restrict__ ob, float* __restrict__ amp) {
  const int r = threadIdx.x;  // 128 threads
  float s = ob[0];
  for (int c = 0; c < MDIM; ++c) s += V[(size_t)r * MDIM + c] * ow[c];
  amp[r] = fabsf(s);
}

// 10) funcs[r][t] = tf_table[ti[r]][t] / (rowmax + 1e-8)
__global__ __launch_bounds__(256) void k_funcs(const float* __restrict__ table,
                                               const int* __restrict__ ti,
                                               float* __restrict__ funcs) {
  const int r = blockIdx.x, tid = threadIdx.x;
  const float* row = table + (size_t)ti[r] * N_SAMP;
  __shared__ float sv[256]; __shared__ float sh;
  float mx = -3.4e38f;
  for (int i = tid; i < N_SAMP; i += 256) mx = fmaxf(mx, row[i]);
  sv[tid] = mx; __syncthreads();
  for (int st = 128; st > 0; st >>= 1) { if (tid < st) sv[tid] = fmaxf(sv[tid], sv[tid + st]); __syncthreads(); }
  if (tid == 0) sh = sv[0];
  __syncthreads();
  const float inv = 1.f / (sh + 1e-8f);
  for (int i = tid; i < N_SAMP; i += 256) funcs[(size_t)r * N_SAMP + i] = row[i] * inv;
}

// 11) env normalize + abs + linear-interp(128->32768) * noise
__global__ __launch_bounds__(256) void k_env(const float* __restrict__ table,
                                             const int* __restrict__ ei,
                                             const float* __restrict__ noise,
                                             float* __restrict__ envN) {
  const int r = blockIdx.x, tid = threadIdx.x;
  const float* row = table + (size_t)ei[r] * ENVF;
  __shared__ float red[256]; __shared__ float e[ENVF]; __shared__ float shm;
  red[tid] = (tid < ENVF) ? row[tid] : -3.4e38f;
  __syncthreads();
  for (int st = 128; st > 0; st >>= 1) { if (tid < st) red[tid] = fmaxf(red[tid], red[tid + st]); __syncthreads(); }
  if (tid == 0) shm = red[0];
  __syncthreads();
  if (tid < ENVF) e[tid] = fabsf(row[tid] / (shm + 1e-8f));
  __syncthreads();
  for (int t = tid; t < N_SAMP; t += 256) {
    float src = (t + 0.5f) * ((float)ENVF / (float)N_SAMP) - 0.5f;
    src = fminf(fmaxf(src, 0.f), (float)(ENVF - 1));
    const int lo = (int)src;  // src >= 0
    const int hi = (lo + 1 < ENVF) ? lo + 1 : ENVF - 1;
    const float w = src - (float)lo;
    const float ev = e[lo] * (1.f - w) + e[hi] * w;
    envN[(size_t)r * N_SAMP + t] = ev * noise[(size_t)r * N_SAMP + t];
  }
}

// 12) real -> complex
__global__ void k_r2c(const float* __restrict__ s, float2* __restrict__ d) {
  const size_t i = (size_t)blockIdx.x * blockDim.x + threadIdx.x;
  float2 v; v.x = s[i]; v.y = 0.f;
  d[i] = v;
}

// 13) one Stockham radix-2 stage over 128 rows of N=32768
__global__ __launch_bounds__(256) void k_fft(const float2* __restrict__ src,
                                             float2* __restrict__ dst,
                                             int l, int m, float sign) {
  const size_t gid = (size_t)blockIdx.x * blockDim.x + threadIdx.x;  // 128*16384
  const int r = (int)(gid >> 14);
  const int k2 = (int)(gid & 16383);
  const int j = k2 / m;
  const int k = k2 - j * m;
  const float2* s = src + (size_t)r * N_SAMP;
  float2* d = dst + (size_t)r * N_SAMP;
  const float2 c0 = s[k + j * m];
  const float2 c1 = s[k + j * m + 16384];  // + l*m (== N/2)
  const float ang = sign * 3.14159265358979323846f * (float)j / (float)l;
  float sw, cw;
  __sincosf(ang, &sw, &cw);
  float2 sum, dif, tw;
  sum.x = c0.x + c1.x; sum.y = c0.y + c1.y;
  dif.x = c0.x - c1.x; dif.y = c0.y - c1.y;
  tw.x = cw * dif.x - sw * dif.y;
  tw.y = cw * dif.y + sw * dif.x;
  d[k + 2 * j * m]     = sum;
  d[k + 2 * j * m + m] = tw;
}

// 14) complex multiply
__global__ void k_cmul(const float2* __restrict__ A, const float2* __restrict__ B,
                       float2* __restrict__ C) {
  const size_t i = (size_t)blockIdx.x * blockDim.x + threadIdx.x;
  const float2 a = A[i], b = B[i];
  float2 c;
  c.x = a.x * b.x - a.y * b.y;
  c.y = a.x * b.y + a.y * b.x;
  C[i] = c;
}

// 15) sig[r][t] = Re(ifft)[r][t] * (1/(N*sqrt(N))) * amp[r]
__global__ void k_extract(const float2* __restrict__ cb, const float* __restrict__ amp,
                          float* __restrict__ sig) {
  const size_t i = (size_t)blockIdx.x * blockDim.x + threadIdx.x;
  const int r = (int)(i >> 15);
  const float scale = 1.0f / (32768.0f * 181.01933598375618f);  // 1/(N*sqrt(N))
  sig[i] = cb[i].x * scale * amp[r];
}

// 16) scatter-add atoms at their time offsets into [B, 2N]
__global__ void k_scatter(const float* __restrict__ sig, const int* __restrict__ idx,
                          float* __restrict__ out2) {
  const size_t i = (size_t)blockIdx.x * blockDim.x + threadIdx.x;  // 128*32768
  const int r = (int)(i >> 15);
  const int t = (int)(i & (N_SAMP - 1));
  const int b = r >> 4, a = r & 15;
  atomicAdd(out2 + (((size_t)b) << 16) + idx[b * NATOMS + a] + t, sig[i]);
}

// 17) final copy of first N samples per batch
__global__ void k_copy(const float* __restrict__ out2, float* __restrict__ out) {
  const size_t i = (size_t)blockIdx.x * blockDim.x + threadIdx.x;  // B*N
  const int b = (int)(i >> 15);
  const int t = (int)(i & (N_SAMP - 1));
  out[i] = out2[(((size_t)b) << 16) + t];
}

// ============================================================
extern "C" void kernel_launch(void* const* d_in, const int* in_sizes, int n_in,
                              void* d_out, int out_size, void* d_ws, size_t ws_size,
                              hipStream_t stream) {
  (void)in_sizes; (void)n_in; (void)out_size; (void)ws_size;
  const float* x        = (const float*)d_in[0];
  const float* fbf      = (const float*)d_in[1];
  const float* dil_w    = (const float*)d_in[2];
  const float* dil_b    = (const float*)d_in[3];
  const float* pw_w     = (const float*)d_in[4];
  const float* pw_b     = (const float*)d_in[5];
  const float* attn_w   = (const float*)d_in[6];
  const float* attn_b   = (const float*)d_in[7];
  const float* tf_table = (const float*)d_in[8];
  const float* env_table= (const float*)d_in[9];
  const float* tf_hw    = (const float*)d_in[10];
  const float* tf_hb    = (const float*)d_in[11];
  const float* tf_ow    = (const float*)d_in[12];
  const float* tf_ob    = (const float*)d_in[13];
  const float* en_hw    = (const float*)d_in[14];
  const float* en_hb    = (const float*)d_in[15];
  const float* en_ow    = (const float*)d_in[16];
  const float* en_ob    = (const float*)d_in[17];
  const float* am_hw    = (const float*)d_in[18];
  const float* am_hb    = (const float*)d_in[19];
  const float* am_ow    = (const float*)d_in[20];
  const float* am_ob    = (const float*)d_in[21];
  const float* noise    = (const float*)d_in[22];
  float* out = (float*)d_out;
  char* base = (char*)d_ws;

  constexpr size_t SZ_H = (size_t)BATCH * MDIM * N_SAMP * sizeof(float);  // 134 MB
  float* H0 = (float*)(base);
  float* H1 = (float*)(base + SZ_H);
  float* Y  = (float*)(base + 2 * SZ_H);
  size_t cur = 3 * SZ_H;
  float* ATTNL = (float*)(base + cur); cur += (size_t)BATCH * N_SAMP * sizeof(float);
  int*   TOPI  = (int*)  (base + cur); cur += 1024;
  float* TOPV  = (float*)(base + cur); cur += 1024;
  float* Z     = (float*)(base + cur); cur += (size_t)MDIM * MDIM * sizeof(float);
  float* VA    = (float*)(base + cur); cur += (size_t)MDIM * MDIM * sizeof(float);
  float* VB    = (float*)(base + cur); cur += (size_t)MDIM * MDIM * sizeof(float);
  float* LTF   = (float*)(base + cur); cur += (size_t)MDIM * NTF * sizeof(float);
  float* LEN   = (float*)(base + cur); cur += (size_t)MDIM * NENV * sizeof(float);
  int*   TI    = (int*)  (base + cur); cur += 1024;
  int*   EI    = (int*)  (base + cur); cur += 1024;
  float* AMP   = (float*)(base + cur); cur += 1024;
  float* OUT2  = (float*)(base + cur); cur += (size_t)BATCH * 2 * N_SAMP * sizeof(float);
  // FFT complex buffers alias H0 (free after conv stack leaves result in H1)
  float2* CB0 = (float2*)(base);
  float2* CB1 = (float2*)(base + SZ_H / 4);
  float2* CB2 = (float2*)(base + SZ_H / 2);
  float2* CB3 = (float2*)(base + 3 * SZ_H / 4);
  // real signal buffers alias Y (free after conv stack)
  float* FUNCS = (float*)((char*)Y);
  float* ENVN  = (float*)((char*)Y + (size_t)MDIM * N_SAMP * sizeof(float));
  float* SIG   = (float*)((char*)Y + 2 * (size_t)MDIM * N_SAMP * sizeof(float));

  const int NT = 256;
  const int GEMM_BLKS = BATCH * N_SAMP / 32;  // 8192

  // --- filterbank conv ---
  k_fbconv<<<GEMM_BLKS, NT, 0, stream>>>(x, fbf, H0);

  // --- dilated residual stack ---
  const int DIL[7] = {1, 3, 9, 27, 81, 243, 1};
  float* hcur = H0; float* hnext = H1;
  for (int i = 0; i < 7; ++i) {
    k_dil<<<GEMM_BLKS, NT, 0, stream>>>(hcur, dil_w + (size_t)i * MDIM * MDIM * 3,
                                        dil_b + (size_t)i * MDIM, Y, DIL[i]);
    k_pw<<<GEMM_BLKS, NT, 0, stream>>>(Y, hcur, pw_w + (size_t)i * MDIM * MDIM,
                                       pw_b + (size_t)i * MDIM, hnext);
    float* t = hcur; hcur = hnext; hnext = t;
  }
  // final h in H1 (7 swaps) -> hcur

  // --- attention, softmax+topk, latent gather ---
  k_attn<<<BATCH * N_SAMP / NT, NT, 0, stream>>>(hcur, attn_w, attn_b, ATTNL);
  k_topk<<<BATCH, NT, 0, stream>>>(ATTNL, TOPI, TOPV);
  k_gatherz<<<MDIM, MDIM, 0, stream>>>(hcur, TOPI, TOPV, Z);

  // --- MLP stacks (WMMA) ---
  // tf
  k_linear<<<4, NT, 0, stream>>>(Z,  tf_hw,          tf_hb,       VA, MDIM, 1);
  k_linear<<<4, NT, 0, stream>>>(VA, tf_hw + 16384,  tf_hb + 128, VB, MDIM, 1);
  k_linear<<<4, NT, 0, stream>>>(VB, tf_hw + 32768,  tf_hb + 256, VA, MDIM, 1);
  k_linear<<<NTF / 32, NT, 0, stream>>>(VA, tf_ow, tf_ob, LTF, NTF, 0);
  // en
  k_linear<<<4, NT, 0, stream>>>(Z,  en_hw,          en_hb,       VA, MDIM, 1);
  k_linear<<<4, NT, 0, stream>>>(VA, en_hw + 16384,  en_hb + 128, VB, MDIM, 1);
  k_linear<<<4, NT, 0, stream>>>(VB, en_hw + 32768,  en_hb + 256, VA, MDIM, 1);
  k_linear<<<NENV / 32, NT, 0, stream>>>(VA, en_ow, en_ob, LEN, NENV, 0);
  // amp
  k_linear<<<4, NT, 0, stream>>>(Z,  am_hw,          am_hb,       VA, MDIM, 1);
  k_linear<<<4, NT, 0, stream>>>(VA, am_hw + 16384,  am_hb + 128, VB, MDIM, 1);
  k_linear<<<4, NT, 0, stream>>>(VB, am_hw + 32768,  am_hb + 256, VA, MDIM, 1);
  k_amp<<<1, MDIM, 0, stream>>>(VA, am_ow, am_ob, AMP);

  k_argmax<<<MDIM, NT, 0, stream>>>(LTF, NTF, TI);
  k_argmax<<<MDIM, NT, 0, stream>>>(LEN, NENV, EI);

  // --- table gathers / normalization / envelope ---
  k_funcs<<<MDIM, NT, 0, stream>>>(tf_table, TI, FUNCS);
  k_env<<<MDIM, NT, 0, stream>>>(env_table, EI, noise, ENVN);

  // --- FFT convolution (Stockham radix-2, 15 stages per direction) ---
  const int FFT_BLKS = MDIM * (N_SAMP / 2) / NT;  // 8192
  const int R2C_BLKS = MDIM * N_SAMP / NT;        // 16384

  k_r2c<<<R2C_BLKS, NT, 0, stream>>>(FUNCS, CB0);
  {
    float2 *pa = CB0, *pb = CB1;
    for (int l = N_SAMP / 2, m = 1; l >= 1; l >>= 1, m <<= 1) {
      k_fft<<<FFT_BLKS, NT, 0, stream>>>(pa, pb, l, m, -1.0f);
      float2* t = pa; pa = pb; pb = t;
    }
    // spectrum(funcs) ends in CB1
  }
  k_r2c<<<R2C_BLKS, NT, 0, stream>>>(ENVN, CB2);
  {
    float2 *pa = CB2, *pb = CB3;
    for (int l = N_SAMP / 2, m = 1; l >= 1; l >>= 1, m <<= 1) {
      k_fft<<<FFT_BLKS, NT, 0, stream>>>(pa, pb, l, m, -1.0f);
      float2* t = pa; pa = pb; pb = t;
    }
    // spectrum(env) ends in CB3
  }
  k_cmul<<<R2C_BLKS, NT, 0, stream>>>(CB1, CB3, CB0);
  float2* ires;
  {
    float2 *pa = CB0, *pb = CB2;
    for (int l = N_SAMP / 2, m = 1; l >= 1; l >>= 1, m <<= 1) {
      k_fft<<<FFT_BLKS, NT, 0, stream>>>(pa, pb, l, m, +1.0f);
      float2* t = pa; pa = pb; pb = t;
    }
    ires = pa;  // CB2
  }
  k_extract<<<R2C_BLKS, NT, 0, stream>>>(ires, AMP, SIG);

  // --- scatter-add and output ---
  hipMemsetAsync(OUT2, 0, (size_t)BATCH * 2 * N_SAMP * sizeof(float), stream);
  k_scatter<<<R2C_BLKS, NT, 0, stream>>>(SIG, TOPI, OUT2);
  k_copy<<<BATCH * N_SAMP / NT, NT, 0, stream>>>(OUT2, out);
}
